// EncoderGRUODE_44598940401831
// MI455X (gfx1250) — matrix operand order repaired and checked by reading, not compiled, not run on hardware
//
#include <hip/hip_runtime.h>

// Problem dims (fixed by the reference)
#define B_   1024
#define T_   200
#define D_   128
#define H_   512     // hidden size = 2*L
#define BT   16      // batch rows per workgroup (WMMA M)
#define NWAVE 8      // waves per block (wave32)
#define WCOLS 64     // output columns per wave (4 WMMA N-tiles)

typedef __attribute__((ext_vector_type(16))) __bf16 v16bf;
typedef __attribute__((ext_vector_type(8)))  __bf16 v8bf;
typedef __attribute__((ext_vector_type(8)))  float  v8f;

static __device__ __forceinline__ v8f zero8() {
  v8f z;
#pragma unroll
  for (int i = 0; i < 8; ++i) z[i] = 0.0f;
  return z;
}

// ---- branch-free activations on native CDNA5 transcendentals ----
static __device__ __forceinline__ float fast_exp2(float x) {
#if __has_builtin(__builtin_amdgcn_exp2f)
  return __builtin_amdgcn_exp2f(x);
#else
  return exp2f(x);
#endif
}
static __device__ __forceinline__ float fast_rcp(float x) {
#if __has_builtin(__builtin_amdgcn_rcpf)
  return __builtin_amdgcn_rcpf(x);
#else
  return 1.0f / x;
#endif
}
static __device__ __forceinline__ float sigmoid_f(float x) {
  x = fminf(fmaxf(x, -30.0f), 30.0f);                     // keep exp2 finite
  return fast_rcp(1.0f + fast_exp2(-1.4426950408889634f * x));
}
static __device__ __forceinline__ float tanh_f(float x) {
#if __has_builtin(__builtin_amdgcn_tanhf)
  return __builtin_amdgcn_tanhf(x);                       // native v_tanh_f32
#else
  x = fminf(fmaxf(x, -15.0f), 15.0f);
  const float e = fast_exp2(2.885390081777927f * x);      // exp(2x)
  return (e - 1.0f) * fast_rcp(e + 1.0f);
#endif
}

// A-fragment (16x32 bf16, M x K) from a row-major bf16 matrix in LDS.
// Lanes 0-15 hold row M=lane with K = k0+{0..15}; lanes 16-31 hold the same
// rows with K = k0+16+{0..15} (matches load_b_bf16).
static __device__ __forceinline__ v16bf load_a_bf16(const __bf16* base, int ld,
                                                    int m, int khalf, int k0) {
  const __bf16* row = base + m * ld;
  const int kb = k0 + khalf * 16;
  v8bf lo = *(const v8bf*)(row + kb);        // K = kb .. kb+7
  v8bf hi = *(const v8bf*)(row + kb + 8);    // K = kb+8 .. kb+15
  return __builtin_shufflevector(lo, hi, 0, 1, 2, 3, 4, 5, 6, 7,
                                 8, 9, 10, 11, 12, 13, 14, 15);
}

// B-fragment (32x16 bf16, K x N) where B(k,n) = W[n0+n][k0+k], W row-major [N,K].
// Lanes 0-15 (N=lane) hold K = k0..k0+15; lanes 16-31 hold K = k0+16..k0+31.
// Each lane reads 32 contiguous bytes -> two global_load_b128 from L2-resident weights.
static __device__ __forceinline__ v16bf load_b_bf16(const __bf16* W, int ldw,
                                                    int n0, int k0, int nlane, int khalf) {
  const __bf16* p = W + (size_t)(n0 + nlane) * (size_t)ldw + k0 + khalf * 16;
  return *(const v16bf*)p;
}

static __device__ __forceinline__ v8f wmma_bf16(v16bf a, v16bf b, v8f c) {
  return __builtin_amdgcn_wmma_f32_16x16x32_bf16(false, a, false, b,
                                                 (short)0, c, false, false);
}

// Dense GEMM helper: 4 N-tiles per wave, K-outer so one A-fragment feeds 4 WMMAs.
static __device__ __forceinline__ void gemm_nt4(const __bf16* A, int lda,
                                                const __bf16* W, int ldw,
                                                int nbase, int kdim, v8f acc[4],
                                                int mrow, int kh, int nl) {
#pragma unroll 2
  for (int k0 = 0; k0 < kdim; k0 += 32) {
    v16bf a = load_a_bf16(A, lda, mrow, kh, k0);
#pragma unroll
    for (int tt = 0; tt < 4; ++tt) {
      v16bf b = load_b_bf16(W, ldw, nbase + 16 * tt, k0, nl, kh);
      acc[tt] = wmma_bf16(a, b, acc[tt]);
    }
  }
}

__global__ void cvt_f32_to_bf16(const float* __restrict__ src,
                                __bf16* __restrict__ dst, int n) {
  int i = blockIdx.x * blockDim.x + threadIdx.x;
  if (i < n) dst[i] = (__bf16)src[i];
}

__global__ __launch_bounds__(BT * NWAVE * 2, 1)
void gruode_scan_kernel(const float* __restrict__ x,      // [B,T,D] f32
                        const float* __restrict__ tps,    // [T] f32
                        const float* __restrict__ b_ih,   // [3H]
                        const float* __restrict__ b_hh,   // [3H]
                        const float* __restrict__ b1,     // [H]
                        const float* __restrict__ b2,     // [H]
                        const float* __restrict__ b_out,  // [H]
                        const __bf16* __restrict__ Wih,   // [3H,D] bf16
                        const __bf16* __restrict__ Whh,   // [3H,H] bf16
                        const __bf16* __restrict__ W1,    // [H,H] bf16
                        const __bf16* __restrict__ W2,    // [H,H] bf16
                        const __bf16* __restrict__ Wout,  // [H,H] bf16
                        float* __restrict__ out)          // [B,H] f32
{
  __shared__ __bf16 hbf[BT][H_];     // hidden state h_t            (16 KB)
  __shared__ __bf16 ubf[BT][H_];     // tanh(h W1^T + b1)           (16 KB)
  __shared__ __bf16 hodebf[BT][H_];  // h + dt * node_f(h)          (16 KB)
  __shared__ __bf16 xtbf[BT][D_];    // x[:, t, :] tile             ( 4 KB)
  __shared__ float  tpsl[T_];        // timestamps                  (0.8 KB)

  const int b0   = blockIdx.x * BT;
  const int tid  = threadIdx.x;
  const int lane = tid & 31;
  const int wave = tid >> 5;
  const int mrow = lane & 15;   // A-fragment row
  const int kh   = lane >> 4;   // K-half select
  const int nl   = lane & 15;   // B/C/D column within tile
  const int nb   = wave * WCOLS;

  // x staging coordinates: 8 contiguous f32 per thread (16*128 = 256*8)
  const int xrow = (tid * 8) >> 7;
  const int xcol = (tid * 8) & 127;
  const float* xbase = x + (size_t)(b0 + xrow) * (size_t)(T_ * D_) + xcol;

  // ---- one-time init: h0 = 0, tps -> LDS, loop-invariant biases -> regs ----
  for (int i = tid; i < BT * H_; i += BT * NWAVE * 2) (&hbf[0][0])[i] = (__bf16)0.0f;
  if (tid < T_) tpsl[tid] = tps[tid];

  float bias1[4], bias2[4];
  float g_bir[4], g_biz[4], g_bin[4], g_bhn[4];   // gate tile q = p*2+tt
#pragma unroll
  for (int q = 0; q < 4; ++q) {
    const int j = nb + 16 * q + nl;
    bias1[q] = b1[j];
    bias2[q] = b2[j];
    g_bir[q] = b_ih[j] + b_hh[j];
    g_biz[q] = b_ih[j + H_] + b_hh[j + H_];
    g_bin[q] = b_ih[j + 2 * H_];
    g_bhn[q] = b_hh[j + 2 * H_];
  }

  // prime the x register pipeline with step 0
  float xreg[8];
#pragma unroll
  for (int e = 0; e < 8; ++e) xreg[e] = xbase[e];

  __syncthreads();

  for (int t = 0; t < T_; ++t) {
    // ---- commit prefetched x[:, t, :] to LDS as bf16 ----
#pragma unroll
    for (int e = 0; e < 8; ++e) xtbf[xrow][xcol + e] = (__bf16)xreg[e];
    const float dt = (t == 0) ? 0.0f : (tpsl[t] - tpsl[t - 1]);
    __syncthreads();

    // ---- prefetch x[:, t+1, :] into registers (overlaps whole step) ----
    if (t + 1 < T_) {
      const float* src = xbase + (size_t)(t + 1) * D_;
#pragma unroll
      for (int e = 0; e < 8; ++e) xreg[e] = src[e];
    }

    // ---- ODE MLP layer 1: u = tanh(h @ W1^T + b1) ----
    {
      v8f acc[4] = {zero8(), zero8(), zero8(), zero8()};
      gemm_nt4(&hbf[0][0], H_, W1, H_, nb, H_, acc, mrow, kh, nl);
#pragma unroll
      for (int tt = 0; tt < 4; ++tt) {
#pragma unroll
        for (int i = 0; i < 8; ++i) {
          const int mm = i + 8 * kh;          // D layout: M = vgpr + 8*(lane>=16)
          ubf[mm][nb + 16 * tt + nl] = (__bf16)tanh_f(acc[tt][i] + bias1[tt]);
        }
      }
    }
    __syncthreads();

    // ---- ODE MLP layer 2 + Euler step: h_ode = h + dt*(u @ W2^T + b2) ----
    {
      v8f acc[4] = {zero8(), zero8(), zero8(), zero8()};
      gemm_nt4(&ubf[0][0], H_, W2, H_, nb, H_, acc, mrow, kh, nl);
#pragma unroll
      for (int tt = 0; tt < 4; ++tt) {
#pragma unroll
        for (int i = 0; i < 8; ++i) {
          const int mm = i + 8 * kh;
          const int j  = nb + 16 * tt + nl;
          const float hv = (float)hbf[mm][j];
          hodebf[mm][j] = (__bf16)(hv + dt * (acc[tt][i] + bias2[tt]));
        }
      }
    }
    __syncthreads();

    // ---- GRU gates: two passes of 2 column-tiles (8 x v8f accs = 64 VGPRs) ----
#pragma unroll 1
    for (int p = 0; p < 2; ++p) {
      const int jb = nb + p * 32;
      v8f ar[2]  = {zero8(), zero8()};
      v8f az[2]  = {zero8(), zero8()};
      v8f ani[2] = {zero8(), zero8()};
      v8f anh[2] = {zero8(), zero8()};

      // input contribution: xt @ Wih^T  (K = 128)
#pragma unroll 2
      for (int k0 = 0; k0 < D_; k0 += 32) {
        v16bf a = load_a_bf16(&xtbf[0][0], D_, mrow, kh, k0);
#pragma unroll
        for (int tt = 0; tt < 2; ++tt) {
          const int n0 = jb + 16 * tt;
          ar[tt]  = wmma_bf16(a, load_b_bf16(Wih, D_, n0,          k0, nl, kh), ar[tt]);
          az[tt]  = wmma_bf16(a, load_b_bf16(Wih, D_, n0 + H_,     k0, nl, kh), az[tt]);
          ani[tt] = wmma_bf16(a, load_b_bf16(Wih, D_, n0 + 2 * H_, k0, nl, kh), ani[tt]);
        }
      }
      // hidden contribution: h_ode @ Whh^T  (K = 512)
#pragma unroll 2
      for (int k0 = 0; k0 < H_; k0 += 32) {
        v16bf a = load_a_bf16(&hodebf[0][0], H_, mrow, kh, k0);
#pragma unroll
        for (int tt = 0; tt < 2; ++tt) {
          const int n0 = jb + 16 * tt;
          ar[tt]  = wmma_bf16(a, load_b_bf16(Whh, H_, n0,          k0, nl, kh), ar[tt]);
          az[tt]  = wmma_bf16(a, load_b_bf16(Whh, H_, n0 + H_,     k0, nl, kh), az[tt]);
          anh[tt] = wmma_bf16(a, load_b_bf16(Whh, H_, n0 + 2 * H_, k0, nl, kh), anh[tt]);
        }
      }
#pragma unroll
      for (int tt = 0; tt < 2; ++tt) {
        const int q = p * 2 + tt;
        const int j = jb + 16 * tt + nl;
#pragma unroll
        for (int i = 0; i < 8; ++i) {
          const int mm = i + 8 * kh;
          const float r  = sigmoid_f(ar[tt][i] + g_bir[q]);
          const float z  = sigmoid_f(az[tt][i] + g_biz[q]);
          const float nn = tanh_f((ani[tt][i] + g_bin[q]) + r * (anh[tt][i] + g_bhn[q]));
          const float ho = (float)hodebf[mm][j];
          hbf[mm][j] = (__bf16)((1.0f - z) * nn + z * ho);
        }
      }
    }
    __syncthreads();   // hbf complete before next step's layer-1 reads it
  }

  // ---- output projection: out = h_T @ Wout^T + b_out ----
  {
    v8f acc[4] = {zero8(), zero8(), zero8(), zero8()};
    gemm_nt4(&hbf[0][0], H_, Wout, H_, nb, H_, acc, mrow, kh, nl);
#pragma unroll
    for (int tt = 0; tt < 4; ++tt) {
      const float bias = b_out[nb + 16 * tt + nl];
#pragma unroll
      for (int i = 0; i < 8; ++i) {
        const int mm = i + 8 * kh;
        out[(size_t)(b0 + mm) * H_ + nb + 16 * tt + nl] = acc[tt][i] + bias;
      }
    }
  }
}

extern "C" void kernel_launch(void* const* d_in, const int* in_sizes, int n_in,
                              void* d_out, int out_size, void* d_ws, size_t ws_size,
                              hipStream_t stream) {
  const float* x     = (const float*)d_in[0];
  const float* tps   = (const float*)d_in[1];
  const float* W_ih  = (const float*)d_in[2];   // [3H, D]
  const float* W_hh  = (const float*)d_in[3];   // [3H, H]
  const float* b_ih  = (const float*)d_in[4];
  const float* b_hh  = (const float*)d_in[5];
  const float* W1    = (const float*)d_in[6];   // [H, H]
  const float* b1    = (const float*)d_in[7];
  const float* W2    = (const float*)d_in[8];   // [H, H]
  const float* b2    = (const float*)d_in[9];
  const float* W_out = (const float*)d_in[10];  // [H, H]
  const float* b_out = (const float*)d_in[11];

  // bf16 weight cache in workspace (~3.4 MB), L2-resident for the whole scan
  __bf16* ws = (__bf16*)d_ws;
  size_t off = 0;
  __bf16* Wih_bf  = ws + off; off += (size_t)3 * H_ * D_;
  __bf16* Whh_bf  = ws + off; off += (size_t)3 * H_ * H_;
  __bf16* W1_bf   = ws + off; off += (size_t)H_ * H_;
  __bf16* W2_bf   = ws + off; off += (size_t)H_ * H_;
  __bf16* Wout_bf = ws + off; off += (size_t)H_ * H_;

  auto cvt = [&](const float* s, __bf16* d, int n) {
    cvt_f32_to_bf16<<<(n + 255) / 256, 256, 0, stream>>>(s, d, n);
  };
  cvt(W_ih,  Wih_bf,  3 * H_ * D_);
  cvt(W_hh,  Whh_bf,  3 * H_ * H_);
  cvt(W1,    W1_bf,   H_ * H_);
  cvt(W2,    W2_bf,   H_ * H_);
  cvt(W_out, Wout_bf, H_ * H_);

  gruode_scan_kernel<<<B_ / BT, BT * NWAVE * 2, 0, stream>>>(
      x, tps, b_ih, b_hh, b1, b2, b_out,
      Wih_bf, Whh_bf, W1_bf, W2_bf, Wout_bf, (float*)d_out);
}
